// Encoder_78520592105658
// MI455X (gfx1250) — matrix operand order repaired and checked by reading, not compiled, not run on hardware
//
#include <hip/hip_runtime.h>
#include <hip/hip_bf16.h>

// ---------- CDNA5 WMMA types ----------
typedef __attribute__((ext_vector_type(16))) __bf16 v16bf;
typedef __attribute__((ext_vector_type(8)))  float  v8f;
typedef __attribute__((ext_vector_type(4)))  float  vf4;

union FragAB { v16bf bf; unsigned int u[8]; unsigned short s[16]; };
union FragC  { v8f f; float x[8]; };
union Pack16 { uint4 q[2]; unsigned short s[16]; };

__device__ __forceinline__ unsigned short f2bf(float f) {
  unsigned int u = __float_as_uint(f);
  unsigned int r = u + 0x7FFFu + ((u >> 16) & 1u);   // round-to-nearest-even
  return (unsigned short)(r >> 16);
}

__device__ __forceinline__ float gelu_f(float u) {
  float t = 0.7978845608028654f * (u + 0.044715f * u * u * u);
  return 0.5f * u * (1.0f + tanhf(t));
}

// ---------- generic bf16 WMMA GEMM: C = alpha*A*B (+bias)(+resid)(gelu) ----------
// A: [M,K] bf16 row stride lda.  B element at B[k*ldbk + n*ldbn] (bf16).
// C/resid at off + row*ldc + col.  Batched over blockIdx.z with (z/ZH, z%ZH) offsets.
template <int BN>
__global__ __launch_bounds__(256) void gemm_bf16_wmma(
    const unsigned short* __restrict__ A, const unsigned short* __restrict__ Bm,
    const float* __restrict__ bias, const float* __restrict__ resid,
    void* __restrict__ Cout, int K,
    long lda, long ldbk, long ldbn, long ldc, int ZH,
    long aB, long aH, long bB, long bH, long cB, long cH,
    float alpha, int flags) {  // flags: 1 = bf16 out, 2 = gelu
  constexpr int BM = 128, BK = 32;
  constexpr int WN = BN / 2;       // wave tile width
  constexpr int NT = WN / 16;      // n subtiles per wave
  constexpr int EB = BK * BN / 256;
  __shared__ unsigned short As[BM][BK + 8];
  __shared__ unsigned short Bs[BN][BK + 8];

  const int tid = threadIdx.x;
  const int wave = tid >> 5, lane = tid & 31;
  const int wm = wave & 3, wn = wave >> 2;
  const int z = blockIdx.z;
  const long offA = (long)(z / ZH) * aB + (long)(z % ZH) * aH;
  const long offB = (long)(z / ZH) * bB + (long)(z % ZH) * bH;
  const long offC = (long)(z / ZH) * cB + (long)(z % ZH) * cH;
  const int bm0 = blockIdx.y * BM;
  const int bn0 = blockIdx.x * BN;

  FragC c[2][4];
#pragma unroll
  for (int mi = 0; mi < 2; ++mi)
#pragma unroll
    for (int ni = 0; ni < 4; ++ni)
#pragma unroll
      for (int r = 0; r < 8; ++r) c[mi][ni].x[r] = 0.0f;

  const int ar = tid >> 1;
  const int ak = (tid & 1) * 16;
  const int tpk = BN / EB;
  const int bk_ = tid / tpk;
  const int bn_ = (tid % tpk) * EB;

#pragma unroll 2
  for (int k0 = 0; k0 < K; k0 += BK) {
    {  // A tile -> LDS (coalesced 32B per thread)
      const unsigned short* src = A + offA + (long)(bm0 + ar) * lda + (k0 + ak);
      *(uint4*)&As[ar][ak] = *(const uint4*)src;
      *(uint4*)&As[ar][ak + 8] = *(const uint4*)(src + 8);
    }
    if (ldbn == 1) {  // B tile -> LDS transposed
      const unsigned short* src = Bm + offB + (long)(k0 + bk_) * ldbk + bn_;
      Pack16 t;
#pragma unroll
      for (int j = 0; j < EB / 8; ++j) t.q[j] = *(const uint4*)(src + 8 * j);
#pragma unroll
      for (int j = 0; j < EB; ++j) Bs[bn_ + j][bk_] = t.s[j];
    } else {
#pragma unroll
      for (int j = 0; j < EB; ++j)
        Bs[bn_ + j][bk_] =
            Bm[offB + (long)(k0 + bk_) * ldbk + (long)(bn_ + j) * ldbn];
    }
    __syncthreads();

    FragAB a[2], fb[4];
    const int kadd = (lane >= 16) ? 16 : 0;
#pragma unroll
    for (int mi = 0; mi < 2; ++mi) {
      const int row = wm * 32 + mi * 16 + (lane & 15);
#pragma unroll
      for (int p = 0; p < 8; ++p) {
        const int kp = 2 * p + kadd;  // contiguous k layout (k = 16h + s)
        a[mi].u[p] = *(const unsigned int*)&As[row][kp];
      }
    }
    const int kb = (lane >= 16) ? 16 : 0;
#pragma unroll
    for (int ni = 0; ni < NT; ++ni) {
      const int col = wn * WN + ni * 16 + (lane & 15);
      uint4 q0 = *(const uint4*)&Bs[col][kb];
      uint4 q1 = *(const uint4*)&Bs[col][kb + 8];
      fb[ni].u[0] = q0.x; fb[ni].u[1] = q0.y; fb[ni].u[2] = q0.z; fb[ni].u[3] = q0.w;
      fb[ni].u[4] = q1.x; fb[ni].u[5] = q1.y; fb[ni].u[6] = q1.z; fb[ni].u[7] = q1.w;
    }
#pragma unroll
    for (int mi = 0; mi < 2; ++mi)
#pragma unroll
      for (int ni = 0; ni < NT; ++ni)
        c[mi][ni].f = __builtin_amdgcn_wmma_f32_16x16x32_bf16(
            false, a[mi].bf, false, fb[ni].bf, (short)0, c[mi][ni].f, false, false);
    __syncthreads();
  }

  const int outbf = flags & 1, dogelu = flags & 2;
#pragma unroll
  for (int mi = 0; mi < 2; ++mi)
#pragma unroll
    for (int ni = 0; ni < NT; ++ni) {
      const int col = bn0 + wn * WN + ni * 16 + (lane & 15);
#pragma unroll
      for (int r = 0; r < 8; ++r) {
        const int row = bm0 + wm * 32 + mi * 16 + r + ((lane >= 16) ? 8 : 0);
        float v = c[mi][ni].x[r] * alpha;
        if (bias) v += bias[col];
        if (resid) v += resid[offC + (long)row * ldc + col];
        if (dogelu) v = gelu_f(v);
        if (outbf)
          ((unsigned short*)Cout)[offC + (long)row * ldc + col] = f2bf(v);
        else
          ((float*)Cout)[offC + (long)row * ldc + col] = v;
      }
    }
}

// ---------- rel scores: for fixed (h,t): S[b,v] = Q8[8x64] . rel[h,t]^T ----------
// streams rel (805MB) exactly once: prefetch + non-temporal loads, WMMA bf16.
__global__ __launch_bounds__(256) void rel_scores_kernel(
    const unsigned short* __restrict__ qkv, const float* __restrict__ rel,
    float* __restrict__ scores) {
  const int t = blockIdx.x;
  const int h = blockIdx.y;
  __shared__ unsigned short qs[16][68];
  const int tid = threadIdx.x;
  for (int i = tid; i < 512; i += 256) {
    const int bb = i >> 6, cc = i & 63;
    qs[bb][cc] = qkv[(long)(bb * 512 + t) * 2304 + h * 64 + cc];
  }
  for (int i = tid; i < 512; i += 256) qs[8 + (i >> 6)][i & 63] = 0;
  __syncthreads();

  const int wave = tid >> 5, lane = tid & 31;
  FragAB a[2];
  {
    const int m = lane & 15;
    const int kadd = (lane >= 16) ? 16 : 0;
#pragma unroll
    for (int ki = 0; ki < 2; ++ki)
#pragma unroll
      for (int p = 0; p < 8; ++p) {
        const int kp = ki * 32 + 2 * p + kadd;
        a[ki].u[p] = *(const unsigned int*)&qs[m][kp];
      }
  }
  const long relbase = (((long)h * 512 + t) * 512) * 64;
  const int kb = (lane >= 16) ? 16 : 0;
#pragma unroll
  for (int ni = 0; ni < 4; ++ni) {
    FragC cc;
#pragma unroll
    for (int r = 0; r < 8; ++r) cc.x[r] = 0.0f;
    const int v = wave * 64 + ni * 16 + (lane & 15);
#pragma unroll
    for (int ki = 0; ki < 2; ++ki) {
      const float* src = rel + relbase + (long)v * 64 + ki * 32 + kb;
      __builtin_prefetch(src + 1024, 0, 0);  // next v-tile, streaming hint
      FragAB fb;
#pragma unroll
      for (int j = 0; j < 4; ++j) {
        vf4 f = __builtin_nontemporal_load((const vf4*)(src + 4 * j));
        fb.s[4 * j + 0] = f2bf(f[0]);
        fb.s[4 * j + 1] = f2bf(f[1]);
        fb.s[4 * j + 2] = f2bf(f[2]);
        fb.s[4 * j + 3] = f2bf(f[3]);
      }
      cc.f = __builtin_amdgcn_wmma_f32_16x16x32_bf16(false, a[ki].bf, false, fb.bf,
                                                     (short)0, cc.f, false, false);
    }
    if (lane < 16) {  // rows 0..7 are real batches; 8..15 are zero-pad
#pragma unroll
      for (int r = 0; r < 8; ++r)
        __builtin_nontemporal_store(
            cc.x[r], &scores[(((long)r * 12 + h) * 512 + t) * 512 + v]);
    }
  }
}

// ---------- fused QK^T * sqrt(HS) + rel -> softmax -> att (bf16) ----------
// Block owns 32 full query rows x all 512 keys for one (b,h); K=64 staged in LDS.
// Saves the 200MB scores round-trip of a separate QK GEMM + softmax.
__global__ __launch_bounds__(256) void qk_softmax_kernel(
    const unsigned short* __restrict__ qkv, const float* __restrict__ relsc,
    unsigned short* __restrict__ att) {
  const int tt = blockIdx.x;        // t tile (32 rows)
  const int z = blockIdx.y;         // b*12 + h
  const int b = z / 12, h = z % 12;
  const int trow0 = tt * 32;
  __shared__ unsigned short As[32][72];   // q tile 32x64 (full K)
  __shared__ unsigned short Bs[512][40];  // k chunk 512x32
  __shared__ float pred[8][32];
  __shared__ float psum[8][32];
  const int tid = threadIdx.x, wave = tid >> 5, lane = tid & 31;

  {  // stage q tile
    const int r = tid >> 3, c8 = (tid & 7) * 8;
    const unsigned short* src =
        qkv + (long)(b * 512 + trow0 + r) * 2304 + h * 64 + c8;
    *(uint4*)&As[r][c8] = *(const uint4*)src;
  }

  FragC c[2][4];
#pragma unroll
  for (int mi = 0; mi < 2; ++mi)
#pragma unroll
    for (int ni = 0; ni < 4; ++ni)
#pragma unroll
      for (int r = 0; r < 8; ++r) c[mi][ni].x[r] = 0.0f;

  const int vrow = tid * 2;
  const int kadd = (lane >= 16) ? 16 : 0;
  const int kb = (lane >= 16) ? 16 : 0;
#pragma unroll
  for (int k0 = 0; k0 < 64; k0 += 32) {
#pragma unroll
    for (int vv = 0; vv < 2; ++vv) {  // stage k chunk (transposed access by c)
      const unsigned short* src =
          qkv + (long)(b * 512 + vrow + vv) * 2304 + 768 + h * 64 + k0;
#pragma unroll
      for (int j = 0; j < 4; ++j)
        *(uint4*)&Bs[vrow + vv][j * 8] = *(const uint4*)(src + j * 8);
    }
    __syncthreads();

    FragAB a[2], fb[4];
#pragma unroll
    for (int mi = 0; mi < 2; ++mi) {
      const int row = mi * 16 + (lane & 15);
#pragma unroll
      for (int p = 0; p < 8; ++p) {
        const int kp = k0 + 2 * p + kadd;
        a[mi].u[p] = *(const unsigned int*)&As[row][kp];
      }
    }
#pragma unroll
    for (int ni = 0; ni < 4; ++ni) {
      const int col = wave * 64 + ni * 16 + (lane & 15);
      uint4 q0 = *(const uint4*)&Bs[col][kb];
      uint4 q1 = *(const uint4*)&Bs[col][kb + 8];
      fb[ni].u[0] = q0.x; fb[ni].u[1] = q0.y; fb[ni].u[2] = q0.z; fb[ni].u[3] = q0.w;
      fb[ni].u[4] = q1.x; fb[ni].u[5] = q1.y; fb[ni].u[6] = q1.z; fb[ni].u[7] = q1.w;
    }
#pragma unroll
    for (int mi = 0; mi < 2; ++mi)
#pragma unroll
      for (int ni = 0; ni < 4; ++ni)
        c[mi][ni].f = __builtin_amdgcn_wmma_f32_16x16x32_bf16(
            false, a[mi].bf, false, fb[ni].bf, (short)0, c[mi][ni].f, false, false);
    __syncthreads();
  }

  // scores = 8*qk + rel  (single read of rel-scores buffer)
  const float* rbase = relsc + ((long)z * 512 + trow0) * 512;
  const int rof = (lane >= 16) ? 8 : 0;
#pragma unroll
  for (int mi = 0; mi < 2; ++mi)
#pragma unroll
    for (int ni = 0; ni < 4; ++ni) {
      const int col = wave * 64 + ni * 16 + (lane & 15);
#pragma unroll
      for (int r = 0; r < 8; ++r) {
        const int row = mi * 16 + r + rof;
        c[mi][ni].x[r] = c[mi][ni].x[r] * 8.0f + rbase[(long)row * 512 + col];
      }
    }

  // row max: lane-local over 4 cols -> xor-reduce over col lanes -> LDS over waves
  float pm[2][8];
#pragma unroll
  for (int mi = 0; mi < 2; ++mi)
#pragma unroll
    for (int r = 0; r < 8; ++r) {
      float m = c[mi][0].x[r];
#pragma unroll
      for (int ni = 1; ni < 4; ++ni) m = fmaxf(m, c[mi][ni].x[r]);
      pm[mi][r] = m;
    }
#pragma unroll
  for (int d = 8; d >= 1; d >>= 1)
#pragma unroll
    for (int mi = 0; mi < 2; ++mi)
#pragma unroll
      for (int r = 0; r < 8; ++r)
        pm[mi][r] = fmaxf(pm[mi][r], __shfl_xor(pm[mi][r], d, 32));
  if ((lane & 15) == 0)
#pragma unroll
    for (int mi = 0; mi < 2; ++mi)
#pragma unroll
      for (int r = 0; r < 8; ++r) pred[wave][mi * 16 + r + rof] = pm[mi][r];
  __syncthreads();

  float rmax[2][8];
#pragma unroll
  for (int mi = 0; mi < 2; ++mi)
#pragma unroll
    for (int r = 0; r < 8; ++r) {
      const int row = mi * 16 + r + rof;
      float m = pred[0][row];
#pragma unroll
      for (int w = 1; w < 8; ++w) m = fmaxf(m, pred[w][row]);
      rmax[mi][r] = m;
    }

  // exp + row sum (same reduction tree)
  float ps[2][8];
#pragma unroll
  for (int mi = 0; mi < 2; ++mi)
#pragma unroll
    for (int r = 0; r < 8; ++r) ps[mi][r] = 0.0f;
#pragma unroll
  for (int mi = 0; mi < 2; ++mi)
#pragma unroll
    for (int ni = 0; ni < 4; ++ni)
#pragma unroll
      for (int r = 0; r < 8; ++r) {
        float e = __expf(c[mi][ni].x[r] - rmax[mi][r]);
        c[mi][ni].x[r] = e;
        ps[mi][r] += e;
      }
#pragma unroll
  for (int d = 8; d >= 1; d >>= 1)
#pragma unroll
    for (int mi = 0; mi < 2; ++mi)
#pragma unroll
      for (int r = 0; r < 8; ++r) ps[mi][r] += __shfl_xor(ps[mi][r], d, 32);
  if ((lane & 15) == 0)
#pragma unroll
    for (int mi = 0; mi < 2; ++mi)
#pragma unroll
      for (int r = 0; r < 8; ++r) psum[wave][mi * 16 + r + rof] = ps[mi][r];
  __syncthreads();

  // normalize -> att bf16
#pragma unroll
  for (int mi = 0; mi < 2; ++mi)
#pragma unroll
    for (int r = 0; r < 8; ++r) {
      const int row = mi * 16 + r + rof;
      float s = 0.0f;
#pragma unroll
      for (int w = 0; w < 8; ++w) s += psum[w][row];
      const float inv = __frcp_rn(s);
#pragma unroll
      for (int ni = 0; ni < 4; ++ni) {
        const int col = wave * 64 + ni * 16 + (lane & 15);
        att[((long)z * 512 + trow0 + row) * 512 + col] = f2bf(c[mi][ni].x[r] * inv);
      }
    }
}

// ---------- RMSNorm row (D=768) -> bf16 ----------
__global__ __launch_bounds__(256) void rmsnorm_kernel(
    const float* __restrict__ x, const float* __restrict__ g,
    unsigned short* __restrict__ out) {
  const long row = blockIdx.x;
  const float* xr = x + row * 768;
  float vals[3], s = 0.0f;
#pragma unroll
  for (int i = 0; i < 3; ++i) {
    vals[i] = xr[threadIdx.x + i * 256];
    s += vals[i] * vals[i];
  }
#pragma unroll
  for (int d = 16; d >= 1; d >>= 1) s += __shfl_xor(s, d, 32);
  __shared__ float red[8];
  if ((threadIdx.x & 31) == 0) red[threadIdx.x >> 5] = s;
  __syncthreads();
  float tot = 0.0f;
#pragma unroll
  for (int i = 0; i < 8; ++i) tot += red[i];
  const float sc = rsqrtf(tot * (1.0f / 768.0f) + 1e-6f);
#pragma unroll
  for (int i = 0; i < 3; ++i) {
    const int col = threadIdx.x + i * 256;
    out[row * 768 + col] = f2bf(vals[i] * sc * g[col]);
  }
}

// ---------- weight casts / packs ----------
__global__ void cast_bf16_kernel(const float* __restrict__ in,
                                 unsigned short* __restrict__ out, long n) {
  const long i = (long)blockIdx.x * 256 + threadIdx.x;
  if (i < n) out[i] = f2bf(in[i]);
}

__global__ void pack_qkv_w_kernel(const float* __restrict__ wq,
                                  const float* __restrict__ wk,
                                  const float* __restrict__ wv,
                                  unsigned short* __restrict__ out) {
  long i = (long)blockIdx.x * 256 + threadIdx.x;  // over 3*12*768*64
  if (i >= 3L * 12 * 768 * 64) return;
  const int c = (int)(i & 63);
  long r = i >> 6;
  const int d = (int)(r % 768);
  r /= 768;
  const int h = (int)(r % 12);
  const int kind = (int)(r / 12);
  const float* w = kind == 0 ? wq : (kind == 1 ? wk : wv);
  out[(long)d * 2304 + kind * 768 + h * 64 + c] = f2bf(w[((long)h * 768 + d) * 64 + c]);
}

__global__ void pack_qkv_b_kernel(const float* __restrict__ bq,
                                  const float* __restrict__ bk,
                                  const float* __restrict__ bv,
                                  float* __restrict__ out) {
  const int i = blockIdx.x * 256 + threadIdx.x;
  if (i < 768) out[i] = bq[i];
  else if (i < 1536) out[i] = bk[i - 768];
  else if (i < 2304) out[i] = bv[i - 1536];
}

// ---------- host orchestration ----------
extern "C" void kernel_launch(void* const* d_in, const int* in_sizes, int n_in,
                              void* d_out, int out_size, void* d_ws, size_t ws_size,
                              hipStream_t stream) {
  (void)in_sizes; (void)n_in; (void)out_size; (void)ws_size;
  const float* x      = (const float*)d_in[0];
  const float* wq     = (const float*)d_in[1];
  const float* bq     = (const float*)d_in[2];
  const float* wk     = (const float*)d_in[3];
  const float* bk     = (const float*)d_in[4];
  const float* wv     = (const float*)d_in[5];
  const float* bv     = (const float*)d_in[6];
  const float* rel    = (const float*)d_in[7];
  const float* w_proj = (const float*)d_in[8];
  const float* b_proj = (const float*)d_in[9];
  const float* w1     = (const float*)d_in[10];
  const float* b1     = (const float*)d_in[11];
  const float* w2     = (const float*)d_in[12];
  const float* b2     = (const float*)d_in[13];
  const float* g      = (const float*)d_in[14];
  float* out = (float*)d_out;

  const long T = 512, D = 768, H = 12, NFF = 3072;
  const long BT = 8 * T;                 // 4096
  const long TT = T * T;                 // 262144

  size_t off = 0;
  char* base = (char*)d_ws;
  auto nxt = [&](size_t bytes) { size_t o = off; off += (bytes + 255) & ~(size_t)255; return o; };
  unsigned short* h_bf   = (unsigned short*)(base + nxt(BT * D * 2));
  unsigned short* wqkv   = (unsigned short*)(base + nxt(D * 3 * D * 2));
  float*          bqkv   = (float*)(base + nxt(3 * D * 4));
  unsigned short* wpj    = (unsigned short*)(base + nxt(D * D * 2));
  unsigned short* w1b    = (unsigned short*)(base + nxt(D * NFF * 2));
  unsigned short* w2b    = (unsigned short*)(base + nxt(NFF * D * 2));
  unsigned short* qkvb   = (unsigned short*)(base + nxt(BT * 3 * D * 2));
  float*          scores = (float*)(base + nxt((size_t)8 * H * TT * 4));
  unsigned short* att    = (unsigned short*)(base + nxt((size_t)8 * H * TT * 2));
  unsigned short* attout = (unsigned short*)(base + nxt(BT * D * 2));
  float*          x1     = (float*)(base + nxt(BT * D * 4));
  unsigned short* h2     = (unsigned short*)(base + nxt(BT * D * 2));
  unsigned short* ubuf   = (unsigned short*)(base + nxt(BT * NFF * 2));

  // weight packing / casting
  pack_qkv_w_kernel<<<(3 * 12 * 768 * 64 + 255) / 256, 256, 0, stream>>>(wq, wk, wv, wqkv);
  pack_qkv_b_kernel<<<9, 256, 0, stream>>>(bq, bk, bv, bqkv);
  cast_bf16_kernel<<<(D * D + 255) / 256, 256, 0, stream>>>(w_proj, wpj, D * D);
  cast_bf16_kernel<<<(D * NFF + 255) / 256, 256, 0, stream>>>(w1, w1b, D * NFF);
  cast_bf16_kernel<<<(NFF * D + 255) / 256, 256, 0, stream>>>(w2, w2b, NFF * D);

  // h = RMSNorm(x)
  rmsnorm_kernel<<<BT, 256, 0, stream>>>(x, g, h_bf);

  // QKV: [4096,768] x [768,2304] + bias -> qkv bf16
  gemm_bf16_wmma<128><<<dim3(18, 32, 1), 256, 0, stream>>>(
      h_bf, wqkv, bqkv, nullptr, qkvb, 768,
      768, 2304, 1, 2304, 1, 0, 0, 0, 0, 0, 0, 1.0f, 1);

  // rel scores -> scores[b,h,t,v]  (streams 805MB of rel exactly once)
  rel_scores_kernel<<<dim3(512, 12, 1), 256, 0, stream>>>(qkvb, rel, scores);

  // fused: att = softmax(sqrt(HS)*QK^T + rel)  -> bf16
  qk_softmax_kernel<<<dim3(16, 96, 1), 256, 0, stream>>>(qkvb, scores, att);

  // out = att @ V  (batched, N=64) -> attout[b,t,h*64+c]
  gemm_bf16_wmma<64><<<dim3(1, 4, 96), 256, 0, stream>>>(
      att, qkvb + 1536, nullptr, nullptr, attout, 512,
      512, 2304, 1, 768, 12,
      H * TT, TT, T * 2304, 64, T * 768, 64, 1.0f, 1);

  // x1 = x + attout @ w_proj + b_proj  (f32)
  gemm_bf16_wmma<128><<<dim3(6, 32, 1), 256, 0, stream>>>(
      attout, wpj, b_proj, x, x1, 768,
      768, 768, 1, 768, 1, 0, 0, 0, 0, 0, 0, 1.0f, 0);

  // h2 = RMSNorm(x1)
  rmsnorm_kernel<<<BT, 256, 0, stream>>>(x1, g, h2);

  // u = gelu(h2 @ w1 + b1) -> bf16
  gemm_bf16_wmma<128><<<dim3(24, 32, 1), 256, 0, stream>>>(
      h2, w1b, b1, nullptr, ubuf, 768,
      768, 3072, 1, 3072, 1, 0, 0, 0, 0, 0, 0, 1.0f, 3);

  // out = x1 + u @ w2 + b2  (f32)
  gemm_bf16_wmma<128><<<dim3(6, 32, 1), 256, 0, stream>>>(
      ubuf, w2b, b2, x1, out, 3072,
      3072, 768, 1, 768, 1, 0, 0, 0, 0, 0, 0, 1.0f, 0);
}
